// Head_82471962018620
// MI455X (gfx1250) — matrix operand order repaired and checked by reading, not compiled, not run on hardware
//
#include <hip/hip_runtime.h>

// ---------------------------------------------------------------------------
// Fast R-CNN head on gfx1250: RoI max-pool -> bf16 activations, then
// WMMA (v_wmma_f32_16x16x32_bf16) GEMMs with on-the-fly fp32->bf16 weight
// conversion during the global->LDS stage (weights are the bandwidth floor:
// W1 fp32 = 411 MB @ 23.3 TB/s ~= 17.6 us; avoid any extra weight passes).
// A-tile staged via GLOBAL_LOAD_ASYNC_TO_LDS_B128 (ASYNCcnt); builtin
// signature (from hipcc diagnostic): (v4i __device__*, v4i __shared__*,
// imm offset, imm cpol) with v4i = int __attribute__((vector_size(16))).
// ---------------------------------------------------------------------------

typedef __attribute__((ext_vector_type(16))) __bf16 v16bf;
typedef __attribute__((ext_vector_type(8)))  float  v8f;

#if defined(__has_builtin)
#  if __has_builtin(__builtin_amdgcn_global_load_async_to_lds_b128) && \
      __has_builtin(__builtin_amdgcn_s_wait_asynccnt)
#    define USE_ASYNC_LDS 1
#  endif
#endif
#ifndef USE_ASYNC_LDS
#  define USE_ASYNC_LDS 0
#endif

#if USE_ASYNC_LDS
typedef int v4i __attribute__((vector_size(4 * sizeof(int))));
typedef __attribute__((address_space(1))) v4i as1_v4i;   // global
typedef __attribute__((address_space(3))) v4i as3_v4i;   // LDS
#endif

#define N_ROIS 512
#define CCH    512
#define FHH    50
#define FWW    50
#define RS     7
#define DFLAT  (CCH * RS * RS)   // 25088
#define FCN    4096
#define SSCALE (1.0f / 16.0f)
#define NEGV   (-1e30f)

__device__ __forceinline__ int imax(int a, int b) { return a > b ? a : b; }
__device__ __forceinline__ int imin(int a, int b) { return a < b ? a : b; }

// round-to-nearest-even float -> bf16 bits
__device__ __forceinline__ unsigned short f2bf(float f) {
    unsigned int u = __float_as_uint(f);
    u += 0x7FFFu + ((u >> 16) & 1u);
    return (unsigned short)(u >> 16);
}

// ---------------------------------------------------------------------------
// Kernel 1: RoI max pool, emitting bf16 activations A[roi][c*49 + i*7 + j].
// x (5.1 MB) lives in L2, so the repeated window reads are L2-bound.
// ---------------------------------------------------------------------------
__global__ __launch_bounds__(256) void roi_pool_bf16(
    const float* __restrict__ x, const float* __restrict__ rois,
    const int* __restrict__ roi_idx, unsigned short* __restrict__ out)
{
    int o = blockIdx.x * 256 + threadIdx.x;
    if (o >= N_ROIS * DFLAT) return;
    int roi = o / DFLAT;
    int rem = o - roi * DFLAT;
    int c   = rem / (RS * RS);
    int ij  = rem - c * (RS * RS);
    int i   = ij / RS;
    int j   = ij - i * RS;

    const float* r = rois + roi * 4;
    int sh = (int)rintf(r[0] * SSCALE);
    int sw = (int)rintf(r[1] * SSCALE);
    int eh = (int)rintf(r[2] * SSCALE);
    int ew = (int)rintf(r[3] * SSCALE);
    float bh = (float)imax(eh - sh + 1, 1) / (float)RS;
    float bw = (float)imax(ew - sw + 1, 1) / (float)RS;

    int hs = imin(imax((int)floorf((float)i * bh) + sh, 0), FHH);
    int he = imin(imax((int)ceilf((float)(i + 1) * bh) + sh, 0), FHH);
    int wS = imin(imax((int)floorf((float)j * bw) + sw, 0), FWW);
    int wE = imin(imax((int)ceilf((float)(j + 1) * bw) + sw, 0), FWW);

    const float* img = x + (size_t)roi_idx[roi] * CCH * FHH * FWW
                         + (size_t)c * FHH * FWW;
    float m = NEGV;
    for (int h = hs; h < he; ++h)
        for (int w = wS; w < wE; ++w)
            m = fmaxf(m, img[h * FWW + w]);
    if (he <= hs || wE <= wS) m = 0.f;
    out[o] = f2bf(m);
}

// ---------------------------------------------------------------------------
// Kernel 2: bf16 WMMA GEMM  C[M,ldc] = act( A[M,K](bf16) x B[K,N](f32) + bias )
// Block: 256 threads = 8 waves (2 m x 4 n). Tile 128x128, BK=64.
// Wave tile 64x32 -> 4x2 accumulators of v8f -> 16 v_wmma per LDS stage.
// A tile: bf16, copied memory->LDS via async-to-LDS (no VGPR round trip).
// B tile: fp32->bf16 converted while staging into LDS transposed [n][k] so
// fragment loads are contiguous 16B ds_load_b128 per the ISA VGPR layouts:
//   A frag: lane row = lane&15, K chunks {kA..kA+7, kA+16..kA+23}, hi half +8
//   B frag: lane col = lane&15, K chunks {kB..kB+7, kB+16..kB+23}, hi half +8
// ---------------------------------------------------------------------------
#define BM 128
#define BN 128
#define BK 64

union FragBF { v16bf v; uint4 u[2]; };

template<bool RELU, bool OUT_BF16, bool FULLN>
__global__ __launch_bounds__(256) void gemm_bias_act(
    const unsigned short* __restrict__ A,   // [M,K] bf16
    const float* __restrict__ B,            // [K,N] f32 row-major
    const float* __restrict__ bias,         // [N]
    void* __restrict__ Cout,                // [M,ldc] bf16 or f32
    int N, int K, int ldc)
{
    __shared__ __align__(16) unsigned short sA[BM * BK];  // [m][k]
    __shared__ __align__(16) unsigned short sB[BN * BK];  // [n][k] (transposed)

    const int tid  = threadIdx.x;
    const int lane = tid & 31;
    const int wv   = tid >> 5;
    const int wm   = (wv >> 2) * 64;   // wave m offset in tile
    const int wn   = (wv & 3) * 32;    // wave n offset in tile
    const int hi   = lane >> 4;        // lane half
    const int lr   = lane & 15;
    const int m0   = blockIdx.y * BM;
    const int n0   = blockIdx.x * BN;

    v8f zero = {0.f, 0.f, 0.f, 0.f, 0.f, 0.f, 0.f, 0.f};
    v8f acc[4][2];
#pragma unroll
    for (int a_ = 0; a_ < 4; ++a_)
#pragma unroll
        for (int b_ = 0; b_ < 2; ++b_) acc[a_][b_] = zero;

    for (int k0 = 0; k0 < K; k0 += BK) {
        // ---- prefetch next k-stage tiles into L2 (global_prefetch_b8) ----
        if (FULLN && (k0 + BK) < K) {
            int pr = tid >> 2;            // 64 weight rows, 4 threads/row
            int pc = (tid & 3) * 32;      // 128B cachelines of fp32
            __builtin_prefetch(&B[(size_t)(k0 + BK + pr) * N + n0 + pc], 0, 1);
            if (tid < BM)                 // one 128B line per activation row
                __builtin_prefetch(&A[(size_t)(m0 + tid) * K + k0 + BK], 0, 1);
        }

        // ---- A tile: 128x64 bf16, async memory->LDS (16B chunks) ----
#pragma unroll
        for (int t = 0; t < 4; ++t) {
            int g   = tid + t * 256;        // 0..1023 groups of 8 elems
            int row = g >> 3;
            int kc  = (g & 7) * 8;
            const unsigned short* gp = &A[(size_t)(m0 + row) * K + (k0 + kc)];
            unsigned short* lp = &sA[row * BK + kc];
#if USE_ASYNC_LDS
            __builtin_amdgcn_global_load_async_to_lds_b128(
                (as1_v4i*)gp, (as3_v4i*)lp, 0, 0);
#else
            *(uint4*)lp = *(const uint4*)gp;
#endif
        }

        // ---- B tile: 64x128 fp32 -> bf16, stored transposed [n][k] ----
#pragma unroll
        for (int t = 0; t < 8; ++t) {
            int g  = tid + t * 256;         // 0..2047 groups of 4 floats
            int kr = g >> 5;                // 0..63
            int nc = (g & 31) * 4;
            if (FULLN) {
                float4 f = *(const float4*)&B[(size_t)(k0 + kr) * N + (n0 + nc)];
                sB[(nc + 0) * BK + kr] = f2bf(f.x);
                sB[(nc + 1) * BK + kr] = f2bf(f.y);
                sB[(nc + 2) * BK + kr] = f2bf(f.z);
                sB[(nc + 3) * BK + kr] = f2bf(f.w);
            } else {
#pragma unroll
                for (int q = 0; q < 4; ++q) {
                    int n = n0 + nc + q;
                    float v = (n < N) ? B[(size_t)(k0 + kr) * N + n] : 0.f;
                    sB[(nc + q) * BK + kr] = f2bf(v);
                }
            }
        }

#if USE_ASYNC_LDS
        __builtin_amdgcn_s_wait_asynccnt(0);   // own async copies landed in LDS
#endif
        __syncthreads();                       // everyone's tiles visible

#pragma unroll
        for (int ks = 0; ks < BK; ks += 32) {
            FragBF fa[4], fb[2];
            int kA = ks + hi * 8;
#pragma unroll
            for (int mt = 0; mt < 4; ++mt) {
                const unsigned short* p = &sA[(wm + mt * 16 + lr) * BK + kA];
                fa[mt].u[0] = *(const uint4*)p;
                fa[mt].u[1] = *(const uint4*)(p + 16);
            }
            int kB = ks + hi * 8;
#pragma unroll
            for (int nt = 0; nt < 2; ++nt) {
                const unsigned short* p = &sB[(wn + nt * 16 + lr) * BK + kB];
                fb[nt].u[0] = *(const uint4*)p;
                fb[nt].u[1] = *(const uint4*)(p + 16);
            }
#pragma unroll
            for (int mt = 0; mt < 4; ++mt)
#pragma unroll
                for (int nt = 0; nt < 2; ++nt)
                    acc[mt][nt] = __builtin_amdgcn_wmma_f32_16x16x32_bf16(
                        false, fa[mt].v, false, fb[nt].v,
                        (short)0, acc[mt][nt], false, false);
        }
        __syncthreads();
    }

    // ---- epilogue: bias (+ReLU), store f32 or bf16 ----
    // C/D layout: VGPR r, lanes 0-15: M=r, N=lane; lanes 16-31: M=8+r.
#pragma unroll
    for (int nt = 0; nt < 2; ++nt) {
        int n = n0 + wn + nt * 16 + lr;
        if (FULLN || n < N) {
            float bv = bias[n];
#pragma unroll
            for (int mt = 0; mt < 4; ++mt) {
#pragma unroll
                for (int r8 = 0; r8 < 8; ++r8) {
                    int m = m0 + wm + mt * 16 + hi * 8 + r8;
                    float v = acc[mt][nt][r8] + bv;
                    if (RELU) v = fmaxf(v, 0.f);
                    if (OUT_BF16)
                        ((unsigned short*)Cout)[(size_t)m * ldc + n] = f2bf(v);
                    else
                        ((float*)Cout)[(size_t)m * ldc + n] = v;
                }
            }
        }
    }
}

// ---------------------------------------------------------------------------
extern "C" void kernel_launch(void* const* d_in, const int* in_sizes, int n_in,
                              void* d_out, int out_size, void* d_ws, size_t ws_size,
                              hipStream_t stream) {
    (void)in_sizes; (void)n_in; (void)out_size; (void)ws_size;
    const float* x     = (const float*)d_in[0];
    const float* rois  = (const float*)d_in[1];
    const int*   ridx  = (const int*)d_in[2];
    const float* W1    = (const float*)d_in[3];
    const float* b1    = (const float*)d_in[4];
    const float* W2    = (const float*)d_in[5];
    const float* b2    = (const float*)d_in[6];
    const float* Wloc  = (const float*)d_in[7];
    const float* bloc  = (const float*)d_in[8];
    const float* Wsc   = (const float*)d_in[9];
    const float* bsc   = (const float*)d_in[10];
    float* out = (float*)d_out;

    // workspace: bf16 pooled activations, fc6 out, fc7 out (~34 MB total)
    unsigned short* A6 = (unsigned short*)d_ws;
    unsigned short* F6 = A6 + (size_t)N_ROIS * DFLAT;
    unsigned short* F7 = F6 + (size_t)N_ROIS * FCN;

    // 1) RoI pool -> bf16 A [512, 25088]
    int total = N_ROIS * DFLAT;
    roi_pool_bf16<<<(total + 255) / 256, 256, 0, stream>>>(x, rois, ridx, A6);

    // 2) fc6: [512,25088] x [25088,4096] + b1, ReLU -> bf16
    dim3 gFC(FCN / BN, N_ROIS / BM);   // (32, 4)
    gemm_bias_act<true, true, true><<<gFC, 256, 0, stream>>>(
        A6, W1, b1, (void*)F6, FCN, DFLAT, FCN);

    // 3) fc7: [512,4096] x [4096,4096] + b2, ReLU -> bf16
    gemm_bias_act<true, true, true><<<gFC, 256, 0, stream>>>(
        F6, W2, b2, (void*)F7, FCN, FCN, FCN);

    // 4) roi_cls_locs: [512,4096] x [4096,84] + b_loc -> f32 @ d_out
    dim3 gS(1, N_ROIS / BM);           // (1, 4)
    gemm_bias_act<false, false, false><<<gS, 256, 0, stream>>>(
        F7, Wloc, bloc, (void*)out, 84, FCN, 84);

    // 5) roi_scores: [512,4096] x [4096,21] + b_score -> f32 @ d_out + 512*84
    gemm_bias_act<false, false, false><<<gS, 256, 0, stream>>>(
        F7, Wsc, bsc, (void*)(out + (size_t)N_ROIS * 84), 21, FCN, 21);
}
